// DecoderTreeRNN_4947802325450
// MI455X (gfx1250) — matrix operand order repaired and checked by reading, not compiled, not run on hardware
//
#include <hip/hip_runtime.h>
#include <hip/hip_bf16.h>
#include <cstdint>

// ---------------------------------------------------------------------------
// DecoderTreeRNN for MI455X (gfx1250): bf16 WMMA GEMMs + VALU gate/softmax.
// B=64, H=1024, V=32000, DEPTH=5.
// GEMM: 64x128 block tile, K staged 64-wide through double-buffered LDS,
// filled with GLOBAL_LOAD_ASYNC_TO_LDS_B128 (ASYNCcnt) when available.
// Each wave: 1 A-frag x 4 B-frags x 2 k-substeps = 8 v_wmma per stage.
// ---------------------------------------------------------------------------

typedef __attribute__((ext_vector_type(16))) __bf16 v16bf;
typedef __attribute__((ext_vector_type(8)))  float  v8f;
typedef int v4i_vec __attribute__((vector_size(16)));   // matches builtin param

#define HD    1024
#define BATCH 64
#define VOC   32000
#define DEPTH 5

struct U32x8 { uint4 lo, hi; };   // 32B carrier for v16bf bit_cast

#if __has_builtin(__builtin_amdgcn_global_load_async_to_lds_b128)
#define USE_ASYNC_LDS 1
#endif

static __device__ __forceinline__ void cp16(const uint16_t* g, uint16_t* l) {
#ifdef USE_ASYNC_LDS
    __builtin_amdgcn_global_load_async_to_lds_b128(
        (__attribute__((address_space(1))) v4i_vec*)g,
        (__attribute__((address_space(3))) v4i_vec*)l, 0, 0);
#else
    *(uint4*)l = *(const uint4*)g;
#endif
}

template <int N>
static __device__ __forceinline__ void wait_async() {
#ifdef USE_ASYNC_LDS
#if __has_builtin(__builtin_amdgcn_s_wait_asynccnt)
    __builtin_amdgcn_s_wait_asynccnt(N);
#else
    asm volatile("s_wait_asynccnt %0" :: "n"(N) : "memory");
#endif
#endif
}

static __device__ __forceinline__ uint16_t f32_to_bf16(float f) {
    uint32_t u = __builtin_bit_cast(uint32_t, f);
    uint32_t r = u + 0x7FFFu + ((u >> 16) & 1u);   // round-to-nearest-even
    return (uint16_t)(r >> 16);
}

__global__ void k_cvt_bf16(const float* __restrict__ src,
                           uint16_t* __restrict__ dst, int n) {
    int i = blockIdx.x * blockDim.x + threadIdx.x;
    int stride = gridDim.x * blockDim.x;
    for (; i < n; i += stride) dst[i] = f32_to_bf16(src[i]);
}

// ---------------------------------------------------------------------------
// C[M,N] = A[M,K] * W[N,K]^T (+ bias).  A,W bf16 (K contiguous), C f32.
// M multiple of 64, N multiple of 128, K multiple of 64.
// 256 threads = 8 waves (wave32): wave (wm, wn0) owns a 16x64 strip.
// ---------------------------------------------------------------------------
__global__ void __launch_bounds__(256)
k_gemm_bf16(const uint16_t* __restrict__ A, const uint16_t* __restrict__ W,
            float* __restrict__ C, const float* __restrict__ bias,
            int M, int N, int K)
{
    // +8 pad -> 144B row stride: conflict-free b128 fragment reads
    __shared__ __align__(16) uint16_t As[2][64][72];
    __shared__ __align__(16) uint16_t Ws[2][128][72];

    const int m0   = blockIdx.y * 64;
    const int n0   = blockIdx.x * 128;
    const int t    = threadIdx.x;
    const int lane = t & 31;
    const int wm   = (t >> 5) & 3;          // wave's 16-row tile (0..3)
    const int wn0  = ((t >> 5) >> 2) * 4;   // wave's first 16-col tile (0 or 4)

    // cooperative staging: each thread moves 16+32 bf16 per stage (6 x b128)
    const int arow = t >> 2;                // 0..63
    const int acol = (t & 3) * 16;          // 0,16,32,48
    const int wrow = t >> 2;                // W rows: wrow and wrow+64

    auto load_stage = [&](int buf, int kb) {
        const uint16_t* gA = A + (size_t)(m0 + arow) * K + kb + acol;
        cp16(gA,     &As[buf][arow][acol]);
        cp16(gA + 8, &As[buf][arow][acol + 8]);
        const uint16_t* gW0 = W + (size_t)(n0 + wrow) * K + kb + acol;
        cp16(gW0,     &Ws[buf][wrow][acol]);
        cp16(gW0 + 8, &Ws[buf][wrow][acol + 8]);
        const uint16_t* gW1 = W + (size_t)(n0 + wrow + 64) * K + kb + acol;
        cp16(gW1,     &Ws[buf][wrow + 64][acol]);
        cp16(gW1 + 8, &Ws[buf][wrow + 64][acol + 8]);
    };

    v8f acc[4] = {};
    const int am = wm * 16 + (lane & 15);

    auto compute_stage = [&](int buf) {
#pragma unroll
        for (int ks = 0; ks < 64; ks += 32) {
            // A fragment, contiguous K layout (matches B):
            //   lanes 0-15 : K=0..15
            //   lanes 16-31: K=16..31
            const int aoff0 = ks + ((lane < 16) ? 0 : 16);
            const int aoff1 = aoff0 + 8;
            U32x8 ta;
            ta.lo = *(const uint4*)&As[buf][am][aoff0];
            ta.hi = *(const uint4*)&As[buf][am][aoff1];
            v16bf af = __builtin_bit_cast(v16bf, ta);
            // B fragment (32x16): lanes 0-15 K=0..15, lanes 16-31 K=16..31
            const int boff = ks + ((lane < 16) ? 0 : 16);
#pragma unroll
            for (int j = 0; j < 4; ++j) {
                const int bn = (wn0 + j) * 16 + (lane & 15);
                U32x8 tb;
                tb.lo = *(const uint4*)&Ws[buf][bn][boff];
                tb.hi = *(const uint4*)&Ws[buf][bn][boff + 8];
                v16bf bf_ = __builtin_bit_cast(v16bf, tb);
                acc[j] = __builtin_amdgcn_wmma_f32_16x16x32_bf16(
                             false, af, false, bf_, (short)0, acc[j], false, false);
            }
        }
    };

    load_stage(0, 0);                        // prologue: stage 0 -> buf 0
    for (int kb = 0; kb < K; kb += 64) {
        const int buf = (kb >> 6) & 1;
        if (kb + 64 < K) {
            load_stage(buf ^ 1, kb + 64);    // overlap next stage's DMA
            wait_async<6>();                 // stage kb done; next 6 in flight
        } else {
            wait_async<0>();
        }
        __syncthreads();                     // all waves' stage-kb data visible
        compute_stage(buf);
        __syncthreads();                     // done reading before overwrite
    }

    // C layout: VGPR v -> row m0+16*wm+v (+8 for lanes 16-31), col = lane&15
    const int crow = m0 + wm * 16 + ((lane < 16) ? 0 : 8);
#pragma unroll
    for (int j = 0; j < 4; ++j) {
        const int cc = n0 + (wn0 + j) * 16 + (lane & 15);
        const float bj = bias ? bias[cc] : 0.0f;
#pragma unroll
        for (int v = 0; v < 8; ++v)
            C[(size_t)(crow + v) * N + cc] = acc[j][v] + bj;
    }
}

// ---------------------------------------------------------------------------
// GRU gating with zero input, scatters children into tree order.
// gh = h @ Whh^T (no bias yet); reference adds bhh then bih inside gates.
// ---------------------------------------------------------------------------
__global__ void k_gru_gate(const float* __restrict__ gh,    // rows x 3H
                           const float* __restrict__ hcur,  // rows x H
                           const float* __restrict__ bih,
                           const float* __restrict__ bhh,
                           float* __restrict__ nextF,       // (2*rows) x H
                           uint16_t* __restrict__ nextB,
                           int rows, int nn, int side)
{
    int idx = blockIdx.x * blockDim.x + threadIdx.x;
    const int total = rows * HD;
    const int stride = gridDim.x * blockDim.x;
    for (; idx < total; idx += stride) {
        const int r = idx >> 10;            // H == 1024
        const int d = idx & (HD - 1);
        const size_t g = (size_t)r * (3 * HD) + d;
        const float ghr = gh[g]          + bhh[d];
        const float ghz = gh[g + HD]     + bhh[HD + d];
        const float ghn = gh[g + 2 * HD] + bhh[2 * HD + d];
        const float rg = 1.0f / (1.0f + __expf(-(bih[d] + ghr)));
        const float zg = 1.0f / (1.0f + __expf(-(bih[HD + d] + ghz)));
        const float ng = tanhf(bih[2 * HD + d] + rg * ghn);
        const float h  = hcur[(size_t)r * HD + d];
        const float o  = (1.0f - zg) * ng + zg * h;
        const int b = r / nn;
        const int i = r - b * nn;
        const size_t crow = (size_t)b * (2 * nn) + 2 * i + side;
        nextF[crow * HD + d] = o;
        nextB[crow * HD + d] = f32_to_bf16(o);
    }
}

// ---------------------------------------------------------------------------
// In-place log_softmax along V per row; one block (256 thr) per row.
// ---------------------------------------------------------------------------
__global__ void __launch_bounds__(256)
k_log_softmax(float* __restrict__ out, int vdim)
{
    __shared__ float red[256];
    float* p = out + (size_t)blockIdx.x * vdim;
    const int t = threadIdx.x;

    float mx = -3.402823466e38f;
    for (int j = t; j < vdim; j += 256) mx = fmaxf(mx, p[j]);
    red[t] = mx; __syncthreads();
    for (int s = 128; s > 0; s >>= 1) {
        if (t < s) red[t] = fmaxf(red[t], red[t + s]);
        __syncthreads();
    }
    mx = red[0]; __syncthreads();

    float sum = 0.0f;
    for (int j = t; j < vdim; j += 256) sum += __expf(p[j] - mx);
    red[t] = sum; __syncthreads();
    for (int s = 128; s > 0; s >>= 1) {
        if (t < s) red[t] += red[t + s];
        __syncthreads();
    }
    const float lz = mx + __logf(red[0]);
    __syncthreads();

    for (int j = t; j < vdim; j += 256) p[j] = p[j] - lz;
}

// ---------------------------------------------------------------------------
extern "C" void kernel_launch(void* const* d_in, const int* in_sizes, int n_in,
                              void* d_out, int out_size, void* d_ws, size_t ws_size,
                              hipStream_t stream)
{
    (void)in_sizes; (void)n_in; (void)out_size; (void)ws_size;
    const float* encoding = (const float*)d_in[0];
    const float* Whh_l    = (const float*)d_in[1];
    const float* bih_l    = (const float*)d_in[2];
    const float* bhh_l    = (const float*)d_in[3];
    const float* Whh_r    = (const float*)d_in[4];
    const float* bih_r    = (const float*)d_in[5];
    const float* bhh_r    = (const float*)d_in[6];
    const float* W_out    = (const float*)d_in[7];
    const float* b_out    = (const float*)d_in[8];
    float* outp = (float*)d_out;

    // ---- workspace carve-out (256B aligned slabs) ----
    uint8_t* ws = (uint8_t*)d_ws;
    size_t off = 0;
    auto carve = [&](size_t bytes) -> void* {
        off = (off + 255) & ~(size_t)255;
        void* p = ws + off;
        off += bytes;
        return p;
    };
    const int NROWS_MAX = BATCH << DEPTH;                       // 2048
    uint16_t* wl   = (uint16_t*)carve((size_t)3 * HD * HD * 2); // Whh_l bf16
    uint16_t* wr   = (uint16_t*)carve((size_t)3 * HD * HD * 2); // Whh_r bf16
    uint16_t* wo   = (uint16_t*)carve((size_t)VOC * HD * 2);    // W_out bf16
    float*    gh   = (float*)   carve((size_t)(BATCH << (DEPTH-1)) * 3 * HD * 4);
    float*    f32b[2];
    uint16_t* bf[2];
    f32b[0] = (float*)   carve((size_t)NROWS_MAX * HD * 4);
    f32b[1] = (float*)   carve((size_t)NROWS_MAX * HD * 4);
    bf[0]   = (uint16_t*)carve((size_t)NROWS_MAX * HD * 2);
    bf[1]   = (uint16_t*)carve((size_t)NROWS_MAX * HD * 2);

    auto cvt = [&](const float* s, uint16_t* d, int n) {
        int blocks = (n + 255) / 256; if (blocks > 8192) blocks = 8192;
        k_cvt_bf16<<<blocks, 256, 0, stream>>>(s, d, n);
    };
    cvt(Whh_l, wl, 3 * HD * HD);
    cvt(Whh_r, wr, 3 * HD * HD);
    cvt(W_out, wo, VOC * HD);
    cvt(encoding, bf[0], BATCH * HD);   // level-0 bf16 state

    const float* curF = encoding;       // level-0 f32 state is the input itself
    uint16_t*    curB = bf[0];

    for (int l = 0; l < DEPTH; ++l) {
        const int rows = BATCH << l;    // 64,128,256,512,1024
        const int nn   = 1 << l;
        float*    nextF = f32b[l & 1];
        uint16_t* nextB = bf[(l + 1) & 1];

        dim3 ggrid(3 * HD / 128, rows / 64);
        int gate_total = rows * HD;
        int gate_blocks = (gate_total + 255) / 256; if (gate_blocks > 4096) gate_blocks = 4096;

        // left child
        k_gemm_bf16<<<ggrid, 256, 0, stream>>>(curB, wl, gh, nullptr, rows, 3 * HD, HD);
        k_gru_gate<<<gate_blocks, 256, 0, stream>>>(gh, curF, bih_l, bhh_l,
                                                    nextF, nextB, rows, nn, 0);
        // right child
        k_gemm_bf16<<<ggrid, 256, 0, stream>>>(curB, wr, gh, nullptr, rows, 3 * HD, HD);
        k_gru_gate<<<gate_blocks, 256, 0, stream>>>(gh, curF, bih_r, bhh_r,
                                                    nextF, nextB, rows, nn, 1);

        curF = nextF;
        curB = nextB;
    }

    // logits = states @ W_out^T + b_out  ->  d_out (2048 x 32000 f32)
    dim3 ogrid(VOC / 128, NROWS_MAX / 64);
    k_gemm_bf16<<<ogrid, 256, 0, stream>>>(curB, wo, outp, b_out,
                                           NROWS_MAX, VOC, HD);

    // log_softmax in place, one block per row
    k_log_softmax<<<NROWS_MAX, 256, 0, stream>>>(outp, VOC);
}
